// GraspSamplerVAE_67190468378988
// MI455X (gfx1250) — matrix operand order repaired and checked by reading, not compiled, not run on hardware
//
#include <hip/hip_runtime.h>
#include <cstdint>

// ============================================================================
// GraspSamplerVAE forward for MI455X (gfx1250), wave32 + WMMA bf16 + async LDS.
//
// Pipeline: FPS -> ball query -> gather -> [WMMA GEMM -> BN stats -> BN+ReLU]x3
//           -> maxpool (x3 SA stages) -> FC x2 -> latent -> decoder (reuses
//           encoder FPS/ball-query geometry) -> heads.
// Conv biases skipped (cancel exactly under train-mode BN).
//
// Input leaf order assumption (insertion-order flattening of setup_inputs()):
//   0 pc 1 grasp 2 features 3 eps
//   enc_p @4 : sa1(l0..2) sa2(l0..2) sa3(l0..2) fc1 fc2, conv = W,b,g,be => 44
//   dec_p @48: same 44, then qW,qb,tW,tb (@92..95)
//   lat_p @96: muW, mub, logvarW, logvarb
// ============================================================================

typedef __bf16 bf16;
typedef __attribute__((ext_vector_type(16))) __bf16 v16bf;
typedef __attribute__((ext_vector_type(8)))  __bf16 v8bf;
typedef __attribute__((ext_vector_type(8)))  float  v8f;

#define BN_EPS 1e-5f
#define NB 8
#define NPTS 32768

// ---------------------------------------------------------------- FPS
__global__ void fps_kernel(const float* __restrict__ xyz, int Np, int S,
                           float* __restrict__ D, int* __restrict__ outIdx,
                           float* __restrict__ centers)
{
  __shared__ float sv[1024];
  __shared__ int   si[1024];
  __shared__ float cpt[3];
  const int b = blockIdx.x, t = threadIdx.x, T = blockDim.x;
  const float* X = xyz + (size_t)b * Np * 3;
  float* Db = D + (size_t)b * Np;
  if (t == 0) {
    outIdx[b * S] = 0;
    cpt[0] = X[0]; cpt[1] = X[1]; cpt[2] = X[2];
    centers[(size_t)b * S * 3 + 0] = X[0];
    centers[(size_t)b * S * 3 + 1] = X[1];
    centers[(size_t)b * S * 3 + 2] = X[2];
  }
  __syncthreads();
  float cx = cpt[0], cy = cpt[1], cz = cpt[2];
  for (int i = t; i < Np; i += T) {
    float dx = X[3*i]-cx, dy = X[3*i+1]-cy, dz = X[3*i+2]-cz;
    Db[i] = dx*dx + dy*dy + dz*dz;
  }
  __syncthreads();
  for (int k = 1; k < S; ++k) {
    float bv = -1.f; int bi = 0x7fffffff;
    for (int i = t; i < Np; i += T) {
      float v = Db[i];
      if (v > bv) { bv = v; bi = i; }
    }
    sv[t] = bv; si[t] = bi;
    __syncthreads();
    for (int o = T >> 1; o > 0; o >>= 1) {
      if (t < o) {
        float v2 = sv[t+o]; int i2 = si[t+o];
        if (v2 > sv[t] || (v2 == sv[t] && i2 < si[t])) { sv[t] = v2; si[t] = i2; }
      }
      __syncthreads();
    }
    int sel = si[0];
    if (t == 0) {
      outIdx[b*S + k] = sel;
      float px = X[3*sel], py = X[3*sel+1], pz = X[3*sel+2];
      cpt[0] = px; cpt[1] = py; cpt[2] = pz;
      centers[((size_t)b*S + k)*3 + 0] = px;
      centers[((size_t)b*S + k)*3 + 1] = py;
      centers[((size_t)b*S + k)*3 + 2] = pz;
    }
    __syncthreads();
    cx = cpt[0]; cy = cpt[1]; cz = cpt[2];
    for (int i = t; i < Np; i += T) {
      float dx = X[3*i]-cx, dy = X[3*i+1]-cy, dz = X[3*i+2]-cz;
      float d2 = dx*dx + dy*dy + dz*dz;
      if (d2 < Db[i]) Db[i] = d2;
    }
    __syncthreads();
  }
}

// ---------------------------------------------------------------- ball query
__global__ void ballquery_kernel(const float* __restrict__ xyz,
                                 const float* __restrict__ centers,
                                 int Np, int S, int ns, float r2,
                                 int* __restrict__ idx)
{
  int gw   = (blockIdx.x * blockDim.x + threadIdx.x) >> 5;
  int lane = threadIdx.x & 31;
  if (gw >= NB * S) return;
  int b = gw / S, s = gw % S;
  const float* X = xyz + (size_t)b * Np * 3;
  const float* c = centers + ((size_t)b * S + s) * 3;
  float cx = c[0], cy = c[1], cz = c[2];
  int* out = idx + ((size_t)b * S + s) * (size_t)ns;
  int count = 0, firstIdx = -1;
  for (int base = 0; base < Np; base += 32) {
    if (count >= ns) break;
    int i = base + lane;
    bool hit = false;
    if (i < Np) {
      float dx = X[3*i]-cx, dy = X[3*i+1]-cy, dz = X[3*i+2]-cz;
      hit = (dx*dx + dy*dy + dz*dz) < r2;
    }
    unsigned m32 = (unsigned)__ballot(hit);
    int prefix = __popc(m32 & ((1u << lane) - 1u));
    if (hit && (count + prefix) < ns) out[count + prefix] = i;
    if (firstIdx < 0 && m32) firstIdx = base + (__ffs(m32) - 1);
    count += __popc(m32);
  }
  if (count == 0) {
    for (int j = lane; j < ns; j += 32) out[j] = 0;
  } else if (count < ns) {
    for (int j = count + lane; j < ns; j += 32) out[j] = firstIdx;
  }
}

// ---------------------------------------------------------------- gathers
__global__ void gather_sa1(const float* __restrict__ pc, const float* __restrict__ extra,
                           int extraC, const float* __restrict__ feat1,
                           const float* __restrict__ centers, const int* __restrict__ idx,
                           bf16* __restrict__ X, int Sl, int ns, int Kpad)
{
  int P = NB * Sl * ns;
  int p = blockIdx.x * blockDim.x + threadIdx.x;
  if (p >= P) return;
  int j = p % ns, s = (p / ns) % Sl, b = p / (ns * Sl);
  int i = idx[((size_t)b * Sl + s) * (size_t)ns + j];
  const float* pt = pc + ((size_t)b * NPTS + i) * 3;
  const float* c  = centers + ((size_t)b * Sl + s) * 3;
  size_t sp = (size_t)P;
  int ch = 0;
  X[(ch++)*sp + p] = (bf16)(pt[0] - c[0]);
  X[(ch++)*sp + p] = (bf16)(pt[1] - c[1]);
  X[(ch++)*sp + p] = (bf16)(pt[2] - c[2]);
  X[(ch++)*sp + p] = (bf16)pt[0];
  X[(ch++)*sp + p] = (bf16)pt[1];
  X[(ch++)*sp + p] = (bf16)pt[2];
  for (int e = 0; e < extraC; ++e) X[(ch++)*sp + p] = (bf16)extra[b*extraC + e];
  X[(ch++)*sp + p] = (bf16)feat1[(size_t)b * NPTS + i];
  for (; ch < Kpad; ++ch) X[(size_t)ch*sp + p] = (bf16)0.f;
}

__global__ void gather_sa2(const float* __restrict__ xyz1, const float* __restrict__ centers2,
                           const int* __restrict__ idx2, const float* __restrict__ F1,
                           bf16* __restrict__ X)
{
  const int Sl = 32, ns = 128, S1 = 128, C = 128, Kpad = 160;
  int P = NB * Sl * ns;
  int p = blockIdx.x * blockDim.x + threadIdx.x;
  if (p >= P) return;
  int j = p % ns, s = (p / ns) % Sl, b = p / (ns * Sl);
  int i = idx2[((size_t)b * Sl + s) * ns + j];
  const float* pt = xyz1 + ((size_t)b * S1 + i) * 3;
  const float* c  = centers2 + ((size_t)b * Sl + s) * 3;
  size_t sp = (size_t)P;
  X[0*sp + p] = (bf16)(pt[0] - c[0]);
  X[1*sp + p] = (bf16)(pt[1] - c[1]);
  X[2*sp + p] = (bf16)(pt[2] - c[2]);
  for (int cc = 0; cc < C; ++cc)
    X[(size_t)(3 + cc)*sp + p] = (bf16)F1[(size_t)cc * (NB*S1) + b*S1 + i];
  for (int ch = 3 + C; ch < Kpad; ++ch) X[(size_t)ch*sp + p] = (bf16)0.f;
}

__global__ void gather_sa3(const float* __restrict__ xyz2, const float* __restrict__ F2,
                           bf16* __restrict__ X)
{
  const int C = 256, Kpad = 288, P = NB * 32;
  int p = blockIdx.x * blockDim.x + threadIdx.x;
  if (p >= P) return;
  int j = p % 32, b = p / 32;
  const float* pt = xyz2 + ((size_t)b * 32 + j) * 3;
  size_t sp = (size_t)P;
  X[0*sp + p] = (bf16)pt[0];
  X[1*sp + p] = (bf16)pt[1];
  X[2*sp + p] = (bf16)pt[2];
  for (int cc = 0; cc < C; ++cc)
    X[(size_t)(3 + cc)*sp + p] = (bf16)F2[(size_t)cc * 256 + b*32 + j];
  for (int ch = 3 + C; ch < Kpad; ++ch) X[(size_t)ch*sp + p] = (bf16)0.f;
}

// ---------------------------------------------------------------- WMMA GEMM
// Y[M][P] = W[M][K] x X[ceil32(K)][P], bf16 inputs, f32 accum.
// 256 threads = 8 waves; block tile 64(M) x 64(N); wave tile 16(M) x 32(N).
// LDS tiles are stored in WMMA *fragment order*:
//   As[m][perm(k)] with perm = swap bits 3<->4 of k  => lane reads 8 at half*8 + 8 at half*8+16
//   Bs[n][k]       (transposed tile)                 => lane reads 16 contiguous
// A tile is DMA'd with global_load_async_to_lds_b128 when rows are 16B-aligned.
__global__ void wmma_gemm(const float* __restrict__ Wf, const bf16* __restrict__ Wb,
                          const bf16* __restrict__ X, float* __restrict__ Y,
                          int M, int K, int P)
{
  __shared__ __attribute__((aligned(16))) bf16 As[64][40];
  __shared__ __attribute__((aligned(16))) bf16 Bs[64][40];
  const int tid = threadIdx.x;
  const int lane = tid & 31, wave = tid >> 5;
  const int wm = wave & 3, wgn = wave >> 2;       // 4 m-tiles x 2 n-groups(32)
  const int m0 = blockIdx.y * 64, n0 = blockIdx.x * 64;
  const int Kp = (K + 31) & ~31;
  const bool asyncA = ((K & 7) == 0);
  const bool fullN  = (n0 + 64 <= P);
  const int half = lane >> 4, mr = lane & 15;
  v8f acc0 = {0.f,0.f,0.f,0.f,0.f,0.f,0.f,0.f};
  v8f acc1 = {0.f,0.f,0.f,0.f,0.f,0.f,0.f,0.f};

  for (int k0 = 0; k0 < Kp; k0 += 32) {
    const bool fast = asyncA && fullN && (k0 + 32 <= K);
    if (fast) {
      // A: one async b128 per thread (64 rows x 4 chunks of 8 bf16).
      //    chunk permutation (swap k bits 3,4) is chunk-contiguous: 0,2,1,3.
      {
        int m = tid >> 2, kc = tid & 3;
        int pc8 = ((kc << 1) & 2) | ((kc >> 1) & 1);
        unsigned lds = (unsigned)(uintptr_t)&As[m][pc8 * 8];
        const bf16* g = Wb + (size_t)(m0 + m) * K + k0 + kc * 8;
        asm volatile("global_load_async_to_lds_b128 %0, %1, off"
                     :: "v"(lds), "v"(g) : "memory");
      }
      // B: 8B coalesced global loads, transposed scatter into LDS.
      #pragma unroll
      for (int it = 0; it < 2; ++it) {
        int i = tid + it * 256;
        int r = i >> 4, c4 = (i & 15) << 2;
        unsigned long long v =
            *(const unsigned long long*)(X + (size_t)(k0 + r) * P + n0 + c4);
        #pragma unroll
        for (int j = 0; j < 4; ++j) {
          unsigned short u = (unsigned short)(v >> (16 * j));
          bf16 bv;
          __builtin_memcpy(&bv, &u, 2);
          Bs[c4 + j][r] = bv;
        }
      }
      asm volatile("s_wait_asynccnt 0x0" ::: "memory");
    } else {
      for (int i = tid; i < 64 * 32; i += 256) {
        int m = i >> 5, k = i & 31;
        int kk = (k & 7) | ((k & 8) << 1) | ((k & 16) >> 1);
        float w = (k0 + k < K) ? Wf[(size_t)(m0 + m) * K + k0 + k] : 0.f;
        As[m][kk] = (bf16)w;
      }
      for (int i = tid; i < 32 * 64; i += 256) {
        int r = i >> 6, c = i & 63;
        int gn = n0 + c;
        Bs[c][r] = (gn < P) ? X[(size_t)(k0 + r) * P + gn] : (bf16)0.f;
      }
    }
    __syncthreads();

    const bf16* arow = &As[wm * 16 + mr][half * 8];
    v8bf alo = *(const v8bf*)(arow);
    v8bf ahi = *(const v8bf*)(arow + 16);
    v16bf af = __builtin_shufflevector(alo, ahi, 0,1,2,3,4,5,6,7,8,9,10,11,12,13,14,15);

    const bf16* b0row = &Bs[wgn * 32 + mr][half * 16];
    v8bf b0l = *(const v8bf*)(b0row);
    v8bf b0h = *(const v8bf*)(b0row + 8);
    v16bf bf0 = __builtin_shufflevector(b0l, b0h, 0,1,2,3,4,5,6,7,8,9,10,11,12,13,14,15);

    const bf16* b1row = &Bs[wgn * 32 + 16 + mr][half * 16];
    v8bf b1l = *(const v8bf*)(b1row);
    v8bf b1h = *(const v8bf*)(b1row + 8);
    v16bf bf1 = __builtin_shufflevector(b1l, b1h, 0,1,2,3,4,5,6,7,8,9,10,11,12,13,14,15);

    acc0 = __builtin_amdgcn_wmma_f32_16x16x32_bf16(false, af, false, bf0,
                                                   (short)0, acc0, false, false);
    acc1 = __builtin_amdgcn_wmma_f32_16x16x32_bf16(false, af, false, bf1,
                                                   (short)0, acc1, false, false);
    __syncthreads();
  }

  // C layout: element e -> m = e + 8*half, n = lane%16
  int n_a = n0 + wgn * 32 + mr;
  int n_b = n_a + 16;
  #pragma unroll
  for (int e = 0; e < 8; ++e) {
    int m = m0 + wm * 16 + e + 8 * half;
    if (n_a < P) Y[(size_t)m * P + n_a] = acc0[e];
    if (n_b < P) Y[(size_t)m * P + n_b] = acc1[e];
  }
}

// ---------------------------------------------------------------- BatchNorm
__global__ void bn_stats(const float* __restrict__ Y, float* __restrict__ mean,
                         float* __restrict__ var, int P)
{
  __shared__ float ss[256], s2[256];
  int m = blockIdx.x, t = threadIdx.x;
  float a = 0.f, b = 0.f;
  for (int i = t; i < P; i += 256) {
    float v = Y[(size_t)m * P + i];
    a += v; b += v * v;
  }
  ss[t] = a; s2[t] = b;
  __syncthreads();
  for (int o = 128; o > 0; o >>= 1) {
    if (t < o) { ss[t] += ss[t+o]; s2[t] += s2[t+o]; }
    __syncthreads();
  }
  if (t == 0) {
    float mu = ss[0] / (float)P;
    mean[m] = mu;
    var[m]  = s2[0] / (float)P - mu * mu;
  }
}

__global__ void bn_relu(const float* __restrict__ Y, const float* __restrict__ mean,
                        const float* __restrict__ var, const float* __restrict__ g,
                        const float* __restrict__ be, bf16* __restrict__ act,
                        int M, int P)
{
  int i = blockIdx.x * blockDim.x + threadIdx.x;
  if (i >= M * P) return;
  int m = i / P;
  float sc = g[m] * rsqrtf(var[m] + BN_EPS);
  float v  = (Y[i] - mean[m]) * sc + be[m];
  act[i] = (bf16)fmaxf(v, 0.f);
}

// ---------------------------------------------------------------- max pool
__global__ void maxpool_kernel(const bf16* __restrict__ act, float* __restrict__ Fo,
                               int C, int BS, int ns)
{
  int o = blockIdx.x * blockDim.x + threadIdx.x;
  if (o >= C * BS) return;
  int c = o / BS, bs = o % BS;
  const bf16* a = act + (size_t)c * BS * ns + (size_t)bs * ns;
  float m = -1e30f;
  for (int j = 0; j < ns; ++j) m = fmaxf(m, (float)a[j]);
  Fo[(size_t)c * BS + bs] = m;
}

__global__ void cvt_kernel(const float* __restrict__ s, bf16* __restrict__ d, int n)
{
  int i = blockIdx.x * blockDim.x + threadIdx.x;
  if (i < n) d[i] = (bf16)s[i];
}

// ---------------------------------------------------------------- latent
__global__ void latent_kernel(const bf16* __restrict__ y, const float* __restrict__ Wmu,
                              const float* __restrict__ bmu, const float* __restrict__ Wlv,
                              const float* __restrict__ blv, const float* __restrict__ eps,
                              float* __restrict__ out, float* __restrict__ z)
{
  int t = threadIdx.x;
  if (t >= 16) return;
  int b = t >> 1, l = t & 1;
  float am = bmu[l], av = blv[l];
  for (int c = 0; c < 1024; ++c) {
    float v = (float)y[c * 8 + b];
    am += v * Wmu[l * 1024 + c];
    av += v * Wlv[l * 1024 + c];
  }
  out[56 + b*2 + l] = am;
  out[72 + b*2 + l] = av;
  z[b*2 + l] = am + eps[b*2 + l] * expf(0.5f * av);
}

// ---------------------------------------------------------------- heads
__global__ void heads_kernel(const bf16* __restrict__ y, const float* __restrict__ Wq,
                             const float* __restrict__ bq, const float* __restrict__ Wt,
                             const float* __restrict__ bt, float* __restrict__ out)
{
  __shared__ float qv[8][4];
  int t = threadIdx.x;
  int b = t / 7, o = t % 7;
  bool valid = t < 56;
  float acc = 0.f;
  if (valid) {
    const float* W = (o < 4) ? Wq : Wt;
    const float* bs = (o < 4) ? bq : bt;
    int oo = (o < 4) ? o : o - 4;
    acc = bs[oo];
    for (int c = 0; c < 1024; ++c) acc += (float)y[c * 8 + b] * W[oo * 1024 + c];
    if (o < 4) qv[b][o] = acc;
  }
  __syncthreads();
  if (valid) {
    if (o < 4) {
      float n2 = qv[b][0]*qv[b][0] + qv[b][1]*qv[b][1] + qv[b][2]*qv[b][2] + qv[b][3]*qv[b][3];
      acc /= fmaxf(sqrtf(n2), 1e-12f);
    }
    out[b * 7 + o] = acc;
  }
}

// ============================================================================
extern "C" void kernel_launch(void* const* d_in, const int* in_sizes, int n_in,
                              void* d_out, int out_size, void* d_ws, size_t ws_size,
                              hipStream_t stream)
{
  const float* pc       = (const float*)d_in[0];
  const float* grasp    = (const float*)d_in[1];
  const float* features = (const float*)d_in[2];
  const float* epsv     = (const float*)d_in[3];
  auto Lf = [&](int i) { return (const float*)d_in[i]; };
  const int ENC = 4, DEC = 48, LAT = 96;
  float* out = (float*)d_out;

  // ---- workspace carve (256B aligned) ----
  char* base = (char*)d_ws; size_t off = 0;
  auto carve = [&](size_t bytes) {
    void* p = base + off;
    off += (bytes + 255) & ~(size_t)255;
    return p;
  };
  float* dist     = (float*)carve((size_t)NB * NPTS * 4);
  float* centers1 = (float*)carve(NB * 128 * 3 * 4);
  int*   fpsIdx1  = (int*)  carve(NB * 128 * 4);
  int*   idx1     = (int*)  carve((size_t)NB * 128 * 64 * 4);
  float* centers2 = (float*)carve(NB * 32 * 3 * 4);
  int*   fpsIdx2  = (int*)  carve(NB * 32 * 4);
  int*   idx2     = (int*)  carve((size_t)NB * 32 * 128 * 4);
  float* pre      = (float*)carve((size_t)8 * 1024 * 1024 * 4);  // max 128x65536 f32
  bf16*  actA     = (bf16*) carve((size_t)8 * 1024 * 1024 * 2);
  bf16*  actB     = (bf16*) carve((size_t)8 * 1024 * 1024 * 2);
  bf16*  wbf      = (bf16*) carve((size_t)1024 * 1024 * 2);      // bf16 weights
  float* F1       = (float*)carve(128 * 1024 * 4);
  float* F2       = (float*)carve(256 * 256 * 4);
  float* F3       = (float*)carve(512 * 8 * 4);
  bf16*  F3b      = (bf16*) carve(512 * 8 * 2);
  float* bn_mean  = (float*)carve(1024 * 4);
  float* bn_var   = (float*)carve(1024 * 4);
  float* zbuf     = (float*)carve(16 * 4);
  (void)ws_size; (void)in_sizes; (void)n_in; (void)out_size;

  // conv layer = [W->bf16] -> WMMA GEMM -> BN stats -> BN+ReLU
  auto layer = [&](int leaf, const bf16* Xin, bf16* Xout, int M, int K, int P) {
    if ((K & 7) == 0) {
      int nw = M * K;
      cvt_kernel<<<(nw + 255) / 256, 256, 0, stream>>>(Lf(leaf + 0), wbf, nw);
    }
    dim3 g((P + 63) / 64, M / 64);
    wmma_gemm<<<g, 256, 0, stream>>>(Lf(leaf + 0), wbf, Xin, pre, M, K, P);
    bn_stats<<<M, 256, 0, stream>>>(pre, bn_mean, bn_var, P);
    int tot = M * P;
    bn_relu<<<(tot + 255) / 256, 256, 0, stream>>>(pre, bn_mean, bn_var,
                                                   Lf(leaf + 2), Lf(leaf + 3),
                                                   Xout, M, P);
  };

  // ---- shared geometry: depends only on pc -> computed once, reused by decoder
  fps_kernel<<<NB, 1024, 0, stream>>>(pc, NPTS, 128, dist, fpsIdx1, centers1);
  ballquery_kernel<<<(NB * 128 + 7) / 8, 256, 0, stream>>>(pc, centers1, NPTS, 128, 64,
                                                           0.02f * 0.02f, idx1);
  fps_kernel<<<NB, 128, 0, stream>>>(centers1, 128, 32, dist, fpsIdx2, centers2);
  ballquery_kernel<<<(NB * 32 + 7) / 8, 256, 0, stream>>>(centers1, centers2, 128, 32, 128,
                                                          0.04f * 0.04f, idx2);

  auto branch = [&](int bl, const float* extra, int extraC) -> bf16* {
    const int P1 = NB * 128 * 64, P2 = NB * 32 * 128, P3 = NB * 32;
    gather_sa1<<<(P1 + 255) / 256, 256, 0, stream>>>(pc, extra, extraC, features,
                                                     centers1, idx1, actA, 128, 64, 32);
    layer(bl + 0, actA, actB, 64, 7 + extraC, P1);   // sa1 l0
    layer(bl + 4, actB, actA, 64, 64, P1);           // sa1 l1
    layer(bl + 8, actA, actB, 128, 64, P1);          // sa1 l2
    maxpool_kernel<<<(128 * 1024 + 255) / 256, 256, 0, stream>>>(actB, F1, 128, 1024, 64);

    gather_sa2<<<(P2 + 255) / 256, 256, 0, stream>>>(centers1, centers2, idx2, F1, actA);
    layer(bl + 12, actA, actB, 128, 131, P2);        // sa2 l0 (K pad 131->160)
    layer(bl + 16, actB, actA, 128, 128, P2);
    layer(bl + 20, actA, actB, 256, 128, P2);
    maxpool_kernel<<<(256 * 256 + 255) / 256, 256, 0, stream>>>(actB, F2, 256, 256, 128);

    gather_sa3<<<1, 256, 0, stream>>>(centers2, F2, actA);
    layer(bl + 24, actA, actB, 256, 259, P3);        // sa3 l0 (259->288)
    layer(bl + 28, actB, actA, 256, 256, P3);
    layer(bl + 32, actA, actB, 512, 256, P3);
    maxpool_kernel<<<(512 * 8 + 255) / 256, 256, 0, stream>>>(actB, F3, 512, 8, 32);

    cvt_kernel<<<(4096 + 255) / 256, 256, 0, stream>>>(F3, F3b, 4096);
    layer(bl + 36, F3b, actA, 1024, 512, 8);         // fc1 (BN1d over batch)
    layer(bl + 40, actA, actB, 1024, 1024, 8);       // fc2
    return actB;                                     // [1024][8] bf16
  };

  bf16* yEnc = branch(ENC, grasp, 7);
  latent_kernel<<<1, 64, 0, stream>>>(yEnc, Lf(LAT + 0), Lf(LAT + 1),
                                      Lf(LAT + 2), Lf(LAT + 3), epsv, out, zbuf);

  bf16* yDec = branch(DEC, zbuf, 2);
  heads_kernel<<<1, 64, 0, stream>>>(yDec, Lf(DEC + 44), Lf(DEC + 45),
                                     Lf(DEC + 46), Lf(DEC + 47), out);
}
